// GNNStack_1434519076933
// MI455X (gfx1250) — matrix operand order repaired and checked by reading, not compiled, not run on hardware
//
#include <hip/hip_runtime.h>
#include <hip/hip_bf16.h>
#include <math.h>

// ---------------------------------------------------------------------------
// GNNStack (2x GraphSage + max-pool + MLP + log_softmax) for gfx1250 (MI455X)
// f16 WMMA (v_wmma_f32_16x16x32_f16) with fp32 accumulate for all GEMMs.
// ---------------------------------------------------------------------------

#define IN_DIM   128
#define HID      128
#define OUT_DIM  2
#define N_GRAPHS 64

typedef _Float16 half16 __attribute__((ext_vector_type(16)));
typedef _Float16 half8  __attribute__((ext_vector_type(8)));
typedef _Float16 half4  __attribute__((ext_vector_type(4)));
typedef float    floatx8 __attribute__((ext_vector_type(8)));

// ---------------------------------------------------------------------------
// One 16x16 output tile accumulated over K with V_WMMA_F32_16X16X32_F16.
//   As : LDS f16 A tile, row-major, ldA halves per row (16B-aligned rows)
//   Wt : global f16 weights, transposed to [n][K] so B loads are contiguous
// A-fragment layout (ISA 7.12.2, 16-bit A 16x32): lane m=lane%16, hs=lane/16:
//   halves 0..7  = A[m][k0 + 8*hs .. +7]
//   halves 8..15 = A[m][k0 + 16 + 8*hs .. +7]
// B-fragment layout (16-bit B 32x16): lane holds column n0+m,
//   halves 0..7 = B[k0 + 8*hs .. +7][n0+m], halves 8..15 = B[k0 + 16 + 8*hs .. +7][n0+m]
// ---------------------------------------------------------------------------
__device__ __forceinline__ floatx8
wmma_row_tile(const _Float16* As, int ldA, const _Float16* __restrict__ Wt,
              int K, int n0, int lane, floatx8 c)
{
    const int m  = lane & 15;
    const int hs = lane >> 4;
    for (int k0 = 0; k0 < K; k0 += 32) {
        half8 a0 = *(const half8*)(As + m * ldA + k0 + hs * 8);
        half8 a1 = *(const half8*)(As + m * ldA + k0 + 16 + hs * 8);
        const half8* pb = (const half8*)(Wt + (size_t)(n0 + m) * K + k0 + hs * 8);
        half8 b0 = pb[0];
        half8 b1 = pb[2];
        half16 a, b;
#pragma unroll
        for (int i = 0; i < 8; ++i) {
            a[i] = a0[i]; a[8 + i] = a1[i];
            b[i] = b0[i]; b[8 + i] = b1[i];
        }
        c = __builtin_amdgcn_wmma_f32_16x16x32_f16(false, a, false, b,
                                                   (short)0, c, false, false);
    }
    return c;
}

// ---------------------------------------------------------------------------
// Y[N,128](f16) = relu(X[N,128](f32) @ W + b), W given transposed f16 [128][128].
// Block: 256 threads = 8 waves; 16 rows x 128 cols per block (1 tile / wave).
// ---------------------------------------------------------------------------
__global__ void linear_relu_msg(const float* __restrict__ X,
                                const _Float16* __restrict__ Wt,
                                const float* __restrict__ bias,
                                _Float16* __restrict__ Y, int N)
{
    __shared__ __attribute__((aligned(16))) _Float16 As[16][136];  // +8 pad
    const int row0 = blockIdx.x * 16;
    const int tid  = threadIdx.x;
    const int wave = tid >> 5;
    const int lane = tid & 31;
    const bool full = (row0 + 16 <= N);   // block-uniform: N % 16 == 0 -> always true

    if (full) {
        for (int i = tid; i < 16 * 128; i += 256) {
            int r = i >> 7, cidx = i & 127;
            As[r][cidx] = (_Float16)X[(size_t)(row0 + r) * 128 + cidx];
        }
    } else {
        for (int i = tid; i < 16 * 128; i += 256) {
            int r = i >> 7, cidx = i & 127;
            int gr = row0 + r;
            As[r][cidx] = (_Float16)((gr < N) ? X[(size_t)gr * 128 + cidx] : 0.0f);
        }
    }
    __syncthreads();

    const int n0 = wave * 16;
    floatx8 acc = {};
    acc = wmma_row_tile(&As[0][0], 136, Wt, 128, n0, lane, acc);

    const int m  = lane & 15;
    const int hs = lane >> 4;
    const int n  = n0 + m;
    const float bn = bias[n];
    if (full) {
#pragma unroll
        for (int r = 0; r < 8; ++r) {
            int gr = row0 + r + hs * 8;
            Y[(size_t)gr * 128 + n] = (_Float16)fmaxf(acc[r] + bn, 0.0f);
        }
    } else {
#pragma unroll
        for (int r = 0; r < 8; ++r) {
            int gr = row0 + r + hs * 8;
            if (gr < N)
                Y[(size_t)gr * 128 + n] = (_Float16)fmaxf(acc[r] + bn, 0.0f);
        }
    }
}

// ---------------------------------------------------------------------------
// Edge scatter: one wave per edge. sum[dst] += msg[src] (fp32 atomics),
// optionally deg[dst] += 1 (degree identical for both layers -> counted once).
// ---------------------------------------------------------------------------
__global__ void edge_scatter(const int* __restrict__ src,
                             const int* __restrict__ dst,
                             const _Float16* __restrict__ msg,
                             float* __restrict__ sum,
                             float* __restrict__ deg, int E)
{
    const int wid  = (blockIdx.x * blockDim.x + threadIdx.x) >> 5;
    const int lane = threadIdx.x & 31;
    if (wid >= E) return;
    const int s = src[wid];
    const int d = dst[wid];
    half4 mv = *(const half4*)(msg + (size_t)s * 128 + lane * 4);
    float* out = sum + (size_t)d * 128 + lane * 4;
    atomicAdd(out + 0, (float)mv[0]);
    atomicAdd(out + 1, (float)mv[1]);
    atomicAdd(out + 2, (float)mv[2]);
    atomicAdd(out + 3, (float)mv[3]);
    if (deg && lane == 0) atomicAdd(deg + d, 1.0f);
}

// ---------------------------------------------------------------------------
// h = l2norm(relu(concat(sum/max(deg,1), X) @ At + b)); At transposed f16 [128][256].
// Optionally stores h (f32) and/or fuses graph max-pool (atomicMax on uint bits,
// valid because h >= 0 and pooled is zero-initialized).
// ---------------------------------------------------------------------------
__global__ void sage_update(const float* __restrict__ sum,
                            const float* __restrict__ deg,
                            const float* __restrict__ X,
                            const _Float16* __restrict__ At,
                            const float* __restrict__ bias,
                            float* __restrict__ Hout,          // may be null
                            const int* __restrict__ batch,     // may be null
                            float* __restrict__ pooled,        // may be null
                            int N)
{
    __shared__ __attribute__((aligned(16))) _Float16 As[16][264];  // 256 + 8 pad
    __shared__ float rowsq[16];
    const int row0 = blockIdx.x * 16;
    const int tid  = threadIdx.x;
    const int wave = tid >> 5;
    const int lane = tid & 31;
    const bool full = (row0 + 16 <= N);

    if (full) {
        for (int i = tid; i < 16 * 256; i += 256) {
            int r = i >> 8, cidx = i & 255;
            int gr = row0 + r;
            float v;
            if (cidx < 128) {
                float dg = fmaxf(deg[gr], 1.0f);
                v = sum[(size_t)gr * 128 + cidx] / dg;
            } else {
                v = X[(size_t)gr * 128 + (cidx - 128)];
            }
            As[r][cidx] = (_Float16)v;
        }
    } else {
        for (int i = tid; i < 16 * 256; i += 256) {
            int r = i >> 8, cidx = i & 255;
            int gr = row0 + r;
            float v = 0.0f;
            if (gr < N) {
                if (cidx < 128) {
                    float dg = fmaxf(deg[gr], 1.0f);
                    v = sum[(size_t)gr * 128 + cidx] / dg;
                } else {
                    v = X[(size_t)gr * 128 + (cidx - 128)];
                }
            }
            As[r][cidx] = (_Float16)v;
        }
    }
    if (tid < 16) rowsq[tid] = 0.0f;
    __syncthreads();

    const int n0 = wave * 16;
    floatx8 acc = {};
    acc = wmma_row_tile(&As[0][0], 264, At, 256, n0, lane, acc);

    const int m  = lane & 15;
    const int hs = lane >> 4;
    const int n  = n0 + m;
    const float bn = bias[n];
    float v[8];
#pragma unroll
    for (int r = 0; r < 8; ++r) {
        v[r] = fmaxf(acc[r] + bn, 0.0f);
        atomicAdd(&rowsq[r + hs * 8], v[r] * v[r]);
    }
    __syncthreads();

#pragma unroll
    for (int r = 0; r < 8; ++r) {
        int M  = r + hs * 8;
        int gr = row0 + M;
        if (full || gr < N) {
            float nrm = sqrtf(rowsq[M]);
            float hv  = v[r] / fmaxf(nrm, 1e-12f);
            if (Hout) Hout[(size_t)gr * 128 + n] = hv;
            if (pooled) {
                int g = batch[gr];
                atomicMax((unsigned int*)&pooled[(size_t)g * 128 + n],
                          __float_as_uint(hv));   // hv >= 0: uint order == float order
            }
        }
    }
}

// ---------------------------------------------------------------------------
// Transpose + fp32->f16: W[K][N] row-major -> Wt[N][K].
// ---------------------------------------------------------------------------
__global__ void prep_weight(const float* __restrict__ W, _Float16* __restrict__ Wt,
                            int K, int Ncols)
{
    int i = blockIdx.x * blockDim.x + threadIdx.x;
    if (i < K * Ncols) {
        int k = i / Ncols, n = i % Ncols;
        Wt[(size_t)n * K + k] = (_Float16)W[(size_t)k * Ncols + n];
    }
}

__global__ void zero_f32(float* __restrict__ p, int n)
{
    int i = blockIdx.x * blockDim.x + threadIdx.x;
    if (i < n) p[i] = 0.0f;
}

// ---------------------------------------------------------------------------
// post_mp: z = (pooled @ w1 + b1) @ w2 + b2; out = log_softmax(z). One block.
// ---------------------------------------------------------------------------
__global__ void post_mlp(const float* __restrict__ pooled,
                         const float* __restrict__ w1, const float* __restrict__ b1,
                         const float* __restrict__ w2, const float* __restrict__ b2,
                         float* __restrict__ out)
{
    __shared__ float t[N_GRAPHS][HID];
    const int tid = threadIdx.x;
    for (int i = tid; i < N_GRAPHS * HID; i += 256) {
        int g = i / HID, j = i % HID;
        float acc = b1[j];
        for (int k = 0; k < HID; ++k)
            acc += pooled[g * HID + k] * w1[k * HID + j];
        t[g][j] = acc;
    }
    __syncthreads();
    if (tid < N_GRAPHS) {
        float z0 = b2[0], z1 = b2[1];
        for (int k = 0; k < HID; ++k) {
            float v = t[tid][k];
            z0 += v * w2[k * OUT_DIM + 0];
            z1 += v * w2[k * OUT_DIM + 1];
        }
        float mx  = fmaxf(z0, z1);
        float lse = mx + logf(expf(z0 - mx) + expf(z1 - mx));
        out[tid * OUT_DIM + 0] = z0 - lse;
        out[tid * OUT_DIM + 1] = z1 - lse;
    }
}

// ---------------------------------------------------------------------------
extern "C" void kernel_launch(void* const* d_in, const int* in_sizes, int n_in,
                              void* d_out, int out_size, void* d_ws, size_t ws_size,
                              hipStream_t stream)
{
    const float* x      = (const float*)d_in[0];
    const int*   eidx   = (const int*)d_in[1];
    const int*   batch  = (const int*)d_in[2];
    const float* lin_w0 = (const float*)d_in[3];
    const float* lin_b0 = (const float*)d_in[4];
    const float* agg_w0 = (const float*)d_in[5];
    const float* agg_b0 = (const float*)d_in[6];
    const float* lin_w1 = (const float*)d_in[7];
    const float* lin_b1 = (const float*)d_in[8];
    const float* agg_w1 = (const float*)d_in[9];
    const float* agg_b1 = (const float*)d_in[10];
    const float* mp_w1  = (const float*)d_in[11];
    const float* mp_b1  = (const float*)d_in[12];
    const float* mp_w2  = (const float*)d_in[13];
    const float* mp_b2  = (const float*)d_in[14];

    const int N = in_sizes[0] / IN_DIM;
    const int E = in_sizes[1] / 2;
    const int* src = eidx;
    const int* dst = eidx + E;

    // ---- workspace partition (256B aligned) ----
    char*  base = (char*)d_ws;
    size_t off  = 0;
    auto alloc = [&](size_t bytes) -> void* {
        void* p = base + off;
        off += (bytes + 255) & ~(size_t)255;
        return p;
    };
    _Float16* w0t    = (_Float16*)alloc((size_t)HID * IN_DIM * 2);          // [128][128]
    _Float16* a0t    = (_Float16*)alloc((size_t)HID * (HID + IN_DIM) * 2);  // [128][256]
    _Float16* w1t    = (_Float16*)alloc((size_t)HID * HID * 2);
    _Float16* a1t    = (_Float16*)alloc((size_t)HID * (2 * HID) * 2);
    _Float16* msg    = (_Float16*)alloc((size_t)N * HID * 2);
    float*    sum    = (float*)alloc((size_t)N * HID * 4);
    float*    deg    = (float*)alloc((size_t)N * 4);
    float*    h0     = (float*)alloc((size_t)N * HID * 4);
    float*    pooled = (float*)alloc((size_t)N_GRAPHS * HID * 4);
    (void)ws_size; (void)n_in;

    const int nodeBlocks = (N + 15) / 16;
    const int edgeBlocks = (E + 7) / 8;   // 8 waves (edges) per 256-thread block

    // 1) weight prep
    prep_weight<<<(128 * 128 + 255) / 256, 256, 0, stream>>>(lin_w0, w0t, 128, 128);
    prep_weight<<<(256 * 128 + 255) / 256, 256, 0, stream>>>(agg_w0, a0t, 256, 128);
    prep_weight<<<(128 * 128 + 255) / 256, 256, 0, stream>>>(lin_w1, w1t, 128, 128);
    prep_weight<<<(256 * 128 + 255) / 256, 256, 0, stream>>>(agg_w1, a1t, 256, 128);

    // 2) zero accumulators
    zero_f32<<<((N * HID) + 255) / 256, 256, 0, stream>>>(sum, N * HID);
    zero_f32<<<(N + 255) / 256, 256, 0, stream>>>(deg, N);
    zero_f32<<<(N_GRAPHS * HID + 255) / 256, 256, 0, stream>>>(pooled, N_GRAPHS * HID);

    // ---- layer 0 ----
    linear_relu_msg<<<nodeBlocks, 256, 0, stream>>>(x, w0t, lin_b0, msg, N);
    edge_scatter<<<edgeBlocks, 256, 0, stream>>>(src, dst, msg, sum, deg, E);
    sage_update<<<nodeBlocks, 256, 0, stream>>>(sum, deg, x, a0t, agg_b0,
                                                h0, nullptr, nullptr, N);

    // ---- layer 1 (pooling fused; h1 never materialized) ----
    zero_f32<<<((N * HID) + 255) / 256, 256, 0, stream>>>(sum, N * HID);
    linear_relu_msg<<<nodeBlocks, 256, 0, stream>>>(h0, w1t, lin_b1, msg, N);
    edge_scatter<<<edgeBlocks, 256, 0, stream>>>(src, dst, msg, sum, nullptr, E);
    sage_update<<<nodeBlocks, 256, 0, stream>>>(sum, deg, h0, a1t, agg_b1,
                                                nullptr, batch, pooled, N);

    // ---- post MLP + log_softmax ----
    post_mlp<<<1, 256, 0, stream>>>(pooled, mp_w1, mp_b1, mp_w2, mp_b2, (float*)d_out);
}
